// SelfAttention_layer_9431748182418
// MI455X (gfx1250) — compile-verified, matrix operand order checked by reading, not run on hardware
//
#include <hip/hip_runtime.h>
#include <stdint.h>

#define DEV __device__ __forceinline__

typedef __attribute__((ext_vector_type(16))) __bf16 v16bf;
typedef __attribute__((ext_vector_type(8)))  float  v8f;
typedef __attribute__((ext_vector_type(4)))  unsigned int v4u;
typedef __attribute__((ext_vector_type(8)))  int v8i_;
typedef __attribute__((ext_vector_type(4)))  int v4i_;

constexpr int BATCH = 4;
constexpr int SEQ   = 2048;
constexpr int DIM   = 1024;
constexpr int NH    = 16;
constexpr int HD    = 64;
constexpr int MTOT  = BATCH * SEQ;   // 8192 rows
constexpr int QKVN  = 3 * DIM;       // 3072

union Frag {
  v16bf v;
  uint4 q[2];
};

DEV unsigned short f2bf(float f) {
  union { float f; uint32_t u; } c; c.f = f;
  uint32_t u = c.u;
  return (unsigned short)((u + 0x7FFFu + ((u >> 16) & 1u)) >> 16);
}

DEV v8f vzero() {
  v8f z;
#pragma unroll
  for (int i = 0; i < 8; i++) z[i] = 0.f;
  return z;
}

DEV v8f bf16_wmma(const Frag& a, const Frag& b, v8f c) {
  return __builtin_amdgcn_wmma_f32_16x16x32_bf16(
      /*neg_a=*/false, a.v, /*neg_b=*/false, b.v,
      /*c_mod=*/(short)0, c, /*reuse_a=*/false, /*reuse_b=*/false);
}

// Load a 16x32 bf16 A-fragment (or 32x16 B-fragment) for one lane.
// base points at (row-of-this-lane, k0 + kbase) in a row-major bf16 matrix.
// Elements 0..7 = K kbase..kbase+7 ; elements 8..15 = K kbase+16..kbase+23.
DEV void load_frag(Frag& f, const unsigned short* base) {
  f.q[0] = *(const uint4*)(base);
  f.q[1] = *(const uint4*)(base + 16);
}

// ---- Tensor Data Mover: DMA a [rows x kelems] bf16 tile (row stride =
// stride_elems) from global into LDS at lds_off. Built per CDNA5 D# spec.
DEV void tdm_load_tile_bf16(const unsigned short* gptr, uint32_t lds_off,
                            int kelems, int rows, int stride_elems,
                            int tensor_d0, int tensor_d1) {
  const uint64_t ga = (uint64_t)(uintptr_t)gptr;
  v4u g0;
  g0[0] = 1u;                                   // count=1 valid descriptor
  g0[1] = lds_off;                              // lds_addr (bytes)
  g0[2] = (uint32_t)ga;                         // global_addr[31:0]
  g0[3] = (uint32_t)((ga >> 32) & 0x1FFFFFFu)   // global_addr[56:32]
          | (2u << 30);                         // type = 2 ("image")
  v8i_ g1;
  g1[0] = (int)(1u << 16);                      // data_size = 1 (2 bytes/elem)
  g1[1] = (int)(((uint32_t)tensor_d0 & 0xFFFFu) << 16);              // tensor_dim0 lo
  g1[2] = (int)((((uint32_t)tensor_d0 >> 16) & 0xFFFFu)
                | (((uint32_t)tensor_d1 & 0xFFFFu) << 16));          // d0 hi | d1 lo
  g1[3] = (int)((((uint32_t)tensor_d1 >> 16) & 0xFFFFu)
                | (((uint32_t)kelems & 0xFFFFu) << 16));             // d1 hi | tile_dim0
  g1[4] = (int)((uint32_t)rows & 0xFFFFu);                           // tile_dim1 (tile_dim2=0)
  g1[5] = (int)(uint32_t)stride_elems;                               // tensor_dim0_stride lo32
  g1[6] = 0;                                                         // stride hi | dim1_stride lo
  g1[7] = 0;
  v4i_ g2 = {0, 0, 0, 0};
  v4i_ g3 = {0, 0, 0, 0};
#if __has_include(<hip/amd_detail/amd_gfx1250_TDM.h>)
  v8i_ g4 = {0, 0, 0, 0, 0, 0, 0, 0};
  __builtin_amdgcn_tensor_load_to_lds(g0, g1, g2, g3, g4, 0);        // clang-23 form
#else
  __builtin_amdgcn_tensor_load_to_lds(g0, g1, g2, g3, 0);            // ROCm 7.2 form
#endif
}

// ---------------- conversion kernels ----------------

__global__ void f32_to_bf16_kernel(const float* __restrict__ in,
                                   unsigned short* __restrict__ out, int n) {
  int i = blockIdx.x * blockDim.x + threadIdx.x;
  if (i < n) out[i] = f2bf(in[i]);
}

// in[rows][cols] fp32 -> out[cols][rows] bf16  (B^T layout for WMMA B-frags)
__global__ void transpose_f32_to_bf16_kernel(const float* __restrict__ in,
                                             unsigned short* __restrict__ out,
                                             int rows, int cols) {
  int c = blockIdx.x * blockDim.x + threadIdx.x;
  int r = blockIdx.y;
  if (c < cols) out[(size_t)c * rows + r] = f2bf(in[(size_t)r * cols + c]);
}

// vT[bh][d][n] = qkv[b*SEQ+n][2*DIM + h*HD + d]
__global__ void vtranspose_kernel(const unsigned short* __restrict__ qkv,
                                  unsigned short* __restrict__ vT) {
  int n  = blockIdx.x * blockDim.x + threadIdx.x;
  int bh = blockIdx.y;
  int b = bh >> 4, h = bh & 15;
  const unsigned short* src = qkv + (size_t)(b * SEQ + n) * QKVN + 2 * DIM + h * HD;
  unsigned short* dst = vT + (size_t)bh * HD * SEQ + n;
#pragma unroll 4
  for (int d = 0; d < HD; d++) dst[(size_t)d * SEQ] = src[d];
}

// ---------------- GEMM: C[M][N] = A[M][K] * Bt[N][K]^T + bias ----------------
// A, Bt bf16 row-major. Block = 4 waves; waves split M (32 rows each), share the
// N0..N0+63 B tile. The shared 64x32 B tile is staged into LDS by the Tensor
// Data Mover (double-buffered), A fragments are direct global_load_b128.

template <bool OUT_BF16>
__global__ __launch_bounds__(128) void gemm_bf16_kernel(
    const unsigned short* __restrict__ A, const unsigned short* __restrict__ Bt,
    const float* __restrict__ bias, void* __restrict__ Cout,
    int M, int N, int K, int ldc) {
  __shared__ unsigned short Bs[2][64 * 32];   // 2 x 4KB double buffer

  const int lane = threadIdx.x & 31;
  const int wave = threadIdx.x >> 5;
  const int m0 = (blockIdx.y * 4 + wave) * 32;
  const int n0 = blockIdx.x * 64;
  const int mrow = lane & 15;         // row (A) / column (B) owned by this lane
  const int kb   = (lane >> 4) * 8;   // per-lane K sub-offset

  v8f acc[2][4];
#pragma unroll
  for (int mi = 0; mi < 2; mi++)
#pragma unroll
    for (int ni = 0; ni < 4; ni++) acc[mi][ni] = vzero();

  const unsigned short* Ar[2];
  Ar[0] = A + (size_t)(m0 + mrow) * K + kb;
  Ar[1] = A + (size_t)(m0 + 16 + mrow) * K + kb;
  const unsigned short* Brow = Bt + (size_t)n0 * K;   // tile base (rows n0..n0+63)

  const int nk = K / 32;
  if (wave == 0) {
    tdm_load_tile_bf16(Brow, (uint32_t)(uintptr_t)&Bs[0][0], 32, 64, K, K, N);
  }

  for (int ki = 0; ki < nk; ki++) {
    const int k0 = ki * 32;
    if (wave == 0) {
      if (ki + 1 < nk) {
        tdm_load_tile_bf16(Brow + k0 + 32,
                           (uint32_t)(uintptr_t)&Bs[(ki + 1) & 1][0],
                           32, 64, K, K, N);
        __builtin_amdgcn_s_wait_tensorcnt(1);   // current buffer landed
      } else {
        __builtin_amdgcn_s_wait_tensorcnt(0);   // last buffer landed
      }
    }
    __syncthreads();   // B tile visible to all waves

    const unsigned short* Bl = &Bs[ki & 1][0];
    Frag a[2], b[4];
    load_frag(a[0], Ar[0] + k0);
    load_frag(a[1], Ar[1] + k0);
#pragma unroll
    for (int ni = 0; ni < 4; ni++) {
      const unsigned short* br = Bl + (ni * 16 + mrow) * 32 + kb;
      b[ni].q[0] = *(const uint4*)(br);
      b[ni].q[1] = *(const uint4*)(br + 16);
    }

    if (k0 + 32 < K) {  // lowers to global_prefetch_b8
      __builtin_prefetch(Ar[0] + k0 + 32, 0, 1);
      __builtin_prefetch(Ar[1] + k0 + 32, 0, 1);
    }

#pragma unroll
    for (int mi = 0; mi < 2; mi++)
#pragma unroll
      for (int ni = 0; ni < 4; ni++)
        acc[mi][ni] = bf16_wmma(a[mi], b[ni], acc[mi][ni]);

    __syncthreads();   // everyone done reading before buffer is re-filled
  }

  const int rsub = (lane >> 4) * 8;  // C/D layout: lanes 16-31 hold rows M+8
#pragma unroll
  for (int ni = 0; ni < 4; ni++) {
    const float bn = bias[n0 + ni * 16 + mrow];
#pragma unroll
    for (int mi = 0; mi < 2; mi++) {
#pragma unroll
      for (int i = 0; i < 8; i++) {
        const int r = m0 + mi * 16 + i + rsub;
        const int cidx = n0 + ni * 16 + mrow;
        const float val = acc[mi][ni][i] + bn;
        if (OUT_BF16)
          ((unsigned short*)Cout)[(size_t)r * ldc + cidx] = f2bf(val);
        else
          ((float*)Cout)[(size_t)r * ldc + cidx] = val;
      }
    }
  }
}

// ---------------- flash attention ----------------
// grid = (SEQ/64, BATCH*NH), block = 128. Each wave: 16 queries x hd=64,
// streaming 32-key blocks with online softmax.

__global__ __launch_bounds__(128) void attn_kernel(
    const unsigned short* __restrict__ qkv,   // [MTOT][3*DIM] bf16
    const unsigned short* __restrict__ vT,    // [B*H][HD][SEQ] bf16
    unsigned short* __restrict__ out) {       // [MTOT][DIM] bf16 (head-merged)
  __shared__ unsigned short Plds[4][16 * 32];

  const int lane = threadIdx.x & 31;
  const int wave = threadIdx.x >> 5;
  const int bh = blockIdx.y;
  const int b = bh >> 4, h = bh & 15;
  const int m0 = blockIdx.x * 64 + wave * 16;
  const int mrow = lane & 15;
  const int kb = (lane >> 4) * 8;
  const int rsub = (lane >> 4) * 8;
  const float scale = 0.125f;  // 64^-0.5

  // Q A-fragments: qf[0] = d 0..31, qf[1] = d 32..63
  Frag qf[2];
  {
    const unsigned short* qb =
        qkv + (size_t)(b * SEQ + m0 + mrow) * QKVN + h * HD;
    load_frag(qf[0], qb + kb);
    load_frag(qf[1], qb + 32 + kb);
  }

  float mrun[8], lrun[8];
#pragma unroll
  for (int i = 0; i < 8; i++) { mrun[i] = -__builtin_inff(); lrun[i] = 0.f; }
  v8f o[4];
#pragma unroll
  for (int dt = 0; dt < 4; dt++) o[dt] = vzero();

  const unsigned short* kmat = qkv + (size_t)(b * SEQ) * QKVN + DIM + h * HD;
  const unsigned short* vb = vT + (size_t)bh * HD * SEQ;
  unsigned short* Pw = Plds[wave];

  for (int key0 = 0; key0 < SEQ; key0 += 32) {
    // ---- S = Q @ K^T  (16q x 32key) ----
    v8f s0 = vzero(), s1 = vzero();
    {
      const unsigned short* kr0 = kmat + (size_t)(key0 + mrow) * QKVN;
      const unsigned short* kr1 = kmat + (size_t)(key0 + 16 + mrow) * QKVN;
      Frag k0a, k0b, k1a, k1b;
      load_frag(k0a, kr0 + kb);        // keys 0-15, d 0..31
      load_frag(k0b, kr0 + 32 + kb);   // keys 0-15, d 32..63
      load_frag(k1a, kr1 + kb);        // keys 16-31
      load_frag(k1b, kr1 + 32 + kb);
      s0 = bf16_wmma(qf[0], k0a, s0);
      s0 = bf16_wmma(qf[1], k0b, s0);
      s1 = bf16_wmma(qf[0], k1a, s1);
      s1 = bf16_wmma(qf[1], k1b, s1);
    }

    // ---- online softmax stats (reductions within 16-lane halves == C rows) ----
    float t[8];
#pragma unroll
    for (int i = 0; i < 8; i++) {
      s0[i] *= scale; s1[i] *= scale;
      t[i] = fmaxf(s0[i], s1[i]);
    }
#pragma unroll
    for (int mask = 1; mask <= 8; mask <<= 1)
#pragma unroll
      for (int i = 0; i < 8; i++)
        t[i] = fmaxf(t[i], __shfl_xor(t[i], mask, 32));

    float sf[8], rs[8];
#pragma unroll
    for (int i = 0; i < 8; i++) {
      const float mn = fmaxf(mrun[i], t[i]);
      sf[i] = __expf(mrun[i] - mn);
      mrun[i] = mn;
      s0[i] = __expf(s0[i] - mn);
      s1[i] = __expf(s1[i] - mn);
      rs[i] = s0[i] + s1[i];
    }
#pragma unroll
    for (int mask = 1; mask <= 8; mask <<= 1)
#pragma unroll
      for (int i = 0; i < 8; i++)
        rs[i] += __shfl_xor(rs[i], mask, 32);
#pragma unroll
    for (int i = 0; i < 8; i++) lrun[i] = lrun[i] * sf[i] + rs[i];
#pragma unroll
    for (int dt = 0; dt < 4; dt++)
#pragma unroll
      for (int i = 0; i < 8; i++) o[dt][i] *= sf[i];

    // ---- C-layout -> A-layout transpose of P through wave-private LDS ----
#pragma unroll
    for (int i = 0; i < 8; i++) {
      Pw[(rsub + i) * 32 + mrow]      = f2bf(s0[i]);
      Pw[(rsub + i) * 32 + 16 + mrow] = f2bf(s1[i]);
    }
    asm volatile("s_wait_dscnt 0" ::: "memory");
    Frag pf;
    pf.q[0] = *(const uint4*)(Pw + mrow * 32 + kb);
    pf.q[1] = *(const uint4*)(Pw + mrow * 32 + kb + 16);

    // ---- O += P @ V  (V supplied transposed: vT[d][n]) ----
#pragma unroll
    for (int dt = 0; dt < 4; dt++) {
      const unsigned short* vr =
          vb + (size_t)(dt * 16 + mrow) * SEQ + key0 + kb;
      Frag vf;
      vf.q[0] = *(const uint4*)(vr);
      vf.q[1] = *(const uint4*)(vr + 16);
      o[dt] = bf16_wmma(pf, vf, o[dt]);
    }
  }

  // ---- epilogue: normalize, write head-merged bf16 ----
#pragma unroll
  for (int i = 0; i < 8; i++) lrun[i] = 1.0f / lrun[i];
#pragma unroll
  for (int dt = 0; dt < 4; dt++)
#pragma unroll
    for (int i = 0; i < 8; i++) {
      const int r = m0 + i + rsub;
      out[(size_t)(b * SEQ + r) * DIM + h * HD + dt * 16 + mrow] =
          f2bf(o[dt][i] * lrun[i]);
    }
}

// ---------------- host launcher ----------------

extern "C" void kernel_launch(void* const* d_in, const int* in_sizes, int n_in,
                              void* d_out, int out_size, void* d_ws, size_t ws_size,
                              hipStream_t stream) {
  const float* x      = (const float*)d_in[0];
  const float* w_qkv  = (const float*)d_in[1];
  const float* b_qkv  = (const float*)d_in[2];
  const float* w_proj = (const float*)d_in[3];
  const float* b_proj = (const float*)d_in[4];
  float* out = (float*)d_out;

  char* ws = (char*)d_ws;
  // xb (16MB) is dead after the QKV GEMM; attn_out (16MB) overlays it.
  unsigned short* xb      = (unsigned short*)(ws);                       // 16 MB
  unsigned short* attnout = xb;                                          // overlay
  unsigned short* wqkvT   = (unsigned short*)(ws + 16777216);            //  6 MB
  unsigned short* wprojT  = (unsigned short*)(ws + 23068672);            //  2 MB
  unsigned short* qkv     = (unsigned short*)(ws + 25165824);            // 48 MB
  unsigned short* vT      = (unsigned short*)(ws + 75497472);            // 16 MB

  // 1) x -> bf16
  {
    int n = MTOT * DIM;
    f32_to_bf16_kernel<<<n / 256, 256, 0, stream>>>(x, xb, n);
  }
  // 2) w_qkv [DIM][3*DIM] -> wqkvT [3*DIM][DIM] bf16
  transpose_f32_to_bf16_kernel<<<dim3(QKVN / 256, DIM), 256, 0, stream>>>(
      w_qkv, wqkvT, DIM, QKVN);
  // 3) w_proj [DIM][DIM] -> wprojT [DIM][DIM] bf16
  transpose_f32_to_bf16_kernel<<<dim3(DIM / 256, DIM), 256, 0, stream>>>(
      w_proj, wprojT, DIM, DIM);
  // 4) qkv = x @ w_qkv + b_qkv   (bf16 out)
  gemm_bf16_kernel<true><<<dim3(QKVN / 64, MTOT / 128), 128, 0, stream>>>(
      xb, wqkvT, b_qkv, (void*)qkv, MTOT, QKVN, DIM, QKVN);
  // 5) v slice -> vT[bh][d][n]
  vtranspose_kernel<<<dim3(SEQ / 256, BATCH * NH), 256, 0, stream>>>(qkv, vT);
  // 6) flash attention -> attnout bf16 [MTOT][DIM]
  attn_kernel<<<dim3(SEQ / 64, BATCH * NH), 128, 0, stream>>>(qkv, vT, attnout);
  // 7) out = attnout @ w_proj + b_proj   (fp32 out)
  gemm_bf16_kernel<false><<<dim3(DIM / 64, MTOT / 128), 128, 0, stream>>>(
      attnout, wprojT, b_proj, (void*)out, MTOT, DIM, DIM, DIM);
}
